// GIN_29076928593943
// MI455X (gfx1250) — matrix operand order repaired and checked by reading, not compiled, not run on hardware
//
#include <hip/hip_runtime.h>

#define NN   100000
#define NE   1600000
#define CH   128
#define OUTC 47
#define BN_EPS 1e-5f

typedef __attribute__((ext_vector_type(16))) __bf16 v16bf;
typedef __attribute__((ext_vector_type(8)))  float  v8f;

union Frag { v16bf v; unsigned short u[16]; };

__device__ __forceinline__ unsigned short f2bf(float f) {
    unsigned int u = __float_as_uint(f);
    unsigned int r = u + 0x7FFFu + ((u >> 16) & 1u);   // round-to-nearest-even
    return (unsigned short)(r >> 16);
}

// ---------------------------------------------------------------- copy (agg = x)
__global__ __launch_bounds__(256) void gin_copy4(const float4* __restrict__ src,
                                                 float4* __restrict__ dst, int n) {
    int i = blockIdx.x * 256 + threadIdx.x;
    if (i < n) dst[i] = src[i];
}

// ---------------------------------------------------------------- edge scatter-add
// agg[dst] += x[src]; 32 lanes per edge, one float4 per lane (128 ch).
__global__ __launch_bounds__(256) void gin_scatter(const float* __restrict__ x,
                                                   const int* __restrict__ ei,
                                                   float* __restrict__ agg) {
    unsigned long long tid = (unsigned long long)blockIdx.x * 256ull + threadIdx.x;
    int e    = (int)(tid >> 5);
    int lane = (int)(tid & 31);
    if (e >= NE) return;
    int dst = ei[e];
    int src = ei[NE + e];
    float4 v = ((const float4*)(x + (size_t)src * CH))[lane];
    float* a = agg + (size_t)dst * CH + lane * 4;
    atomicAdd(a + 0, v.x);
    atomicAdd(a + 1, v.y);
    atomicAdd(a + 2, v.z);
    atomicAdd(a + 3, v.w);
}

// ---------------------------------------------------------------- fused GIN MLP
// block = 256 threads (8 waves), 128 rows/block. Per wave: 16 rows.
// GEMM1 (128x128, bf16 WMMA) -> bias+ReLU+BN affine -> LDS -> GEMM2 (128xDOUT)
// -> (ReLU | fused log_softmax).
template <int DOUT, bool LAST>
__global__ __launch_bounds__(256) void gin_mlp(
    const float* __restrict__ h,       // [NN][128]  (x + agg)
    const float* __restrict__ W1,      // [128][128]
    const float* __restrict__ b1,      // [128]
    const float* __restrict__ gamma, const float* __restrict__ beta,
    const float* __restrict__ mean,  const float* __restrict__ var,
    const float* __restrict__ W2,      // [128][DOUT]
    const float* __restrict__ b2,      // [DOUT]
    float* __restrict__ out)           // [NN][DOUT]
{
    constexpr int NT2    = (DOUT + 15) / 16;
    constexpr int W1_U16 = 4 * 8 * 32 * 16;     // 32 frags * 512 bf16
    constexpr int W2_U16 = 4 * NT2 * 32 * 16;
    constexpr int T_U16  = 8 * 4 * 32 * 16;

    __shared__ __align__(32) unsigned short sW1[W1_U16];
    __shared__ __align__(32) unsigned short sW2[W2_U16];
    __shared__ __align__(32) unsigned short sT[T_U16];
    __shared__ float sB1[128], sScale[128], sShift[128], sB2[NT2 * 16];
    float* sLogit = (float*)sW1;   // overlay: W1 frags dead after mid-barrier

    const int tid  = threadIdx.x;
    const int wave = tid >> 5;
    const int lane = tid & 31;

    // zero-pad W2 fragment region (covers DOUT % 16 != 0)
    for (int i = tid; i < W2_U16; i += 256) sW2[i] = 0;
    __syncthreads();

    // stage W1 as bf16 in WMMA B-fragment layout
    for (int i = tid; i < 128 * 128; i += 256) {
        int k = i >> 7, n = i & 127;
        int kt = k >> 5, kk = k & 31;
        int frag = kt * 8 + (n >> 4);
        int ln   = (((kk >> 3) & 1) ? 16 : 0) + (n & 15);
        sW1[(frag * 32 + ln) * 16 + ((kk & 7) + ((kk >> 4) << 3))] = f2bf(W1[i]);
    }
    // stage W2 likewise
    for (int i = tid; i < 128 * DOUT; i += 256) {
        int k = i / DOUT, n = i - k * DOUT;
        int kt = k >> 5, kk = k & 31;
        int frag = kt * NT2 + (n >> 4);
        int ln   = (((kk >> 3) & 1) ? 16 : 0) + (n & 15);
        sW2[(frag * 32 + ln) * 16 + ((kk & 7) + ((kk >> 4) << 3))] = f2bf(W2[i]);
    }
    if (tid < 128) {
        float sc = gamma[tid] * rsqrtf(var[tid] + BN_EPS);
        sScale[tid] = sc;
        sShift[tid] = beta[tid] - mean[tid] * sc;
        sB1[tid]    = b1[tid];
    }
    if (tid < NT2 * 16) sB2[tid] = (tid < DOUT) ? b2[tid] : 0.f;
    __syncthreads();

    const int m0    = blockIdx.x * 128 + wave * 16;
    const int mrow  = m0 + (lane & 15);
    const int koff  = (lane < 16) ? 0 : 8;
    const int cLo   = lane & 15;
    const int mbase = (lane < 16) ? 0 : 8;

    // A fragments for GEMM1 (whole K=128 per wave-tile)
    Frag A[4];
#pragma unroll
    for (int kt = 0; kt < 4; ++kt) {
        if (mrow < NN) {
            const float* p = h + (size_t)mrow * CH + kt * 32 + koff;
#pragma unroll
            for (int j = 0; j < 8; ++j) A[kt].u[j]     = f2bf(p[j]);
#pragma unroll
            for (int j = 0; j < 8; ++j) A[kt].u[8 + j] = f2bf(p[16 + j]);
        } else {
#pragma unroll
            for (int j = 0; j < 16; ++j) A[kt].u[j] = 0;
        }
    }

    // ---- GEMM1 + bias + ReLU + BN, scatter into A-fragment layout of sT ----
#pragma unroll
    for (int nt = 0; nt < 8; ++nt) {
        v8f acc = {};
#pragma unroll
        for (int kt = 0; kt < 4; ++kt) {
            Frag B;
            B.v = *(const v16bf*)&sW1[((kt * 8 + nt) * 32 + lane) * 16];
            acc = __builtin_amdgcn_wmma_f32_16x16x32_bf16(
                false, A[kt].v, false, B.v, (short)0, acc, false, false);
        }
        int col = nt * 16 + cLo;
        float sc = sScale[col], sh = sShift[col], bb = sB1[col];
        int kk   = col & 31;
        int hsel = (kk >> 3) & 1;
        int e    = (kk & 7) + ((kk >> 4) << 3);
        int ktc  = col >> 5;
#pragma unroll
        for (int r = 0; r < 8; ++r) {
            int M   = mbase + r;
            float z = fmaxf(acc[r] + bb, 0.f);
            z = z * sc + sh;
            sT[((wave * 4 + ktc) * 32 + hsel * 16 + M) * 16 + e] = f2bf(z);
        }
    }
    __syncthreads();

    // ---- GEMM2 ----
    Frag A2[4];
#pragma unroll
    for (int kt = 0; kt < 4; ++kt)
        A2[kt].v = *(const v16bf*)&sT[((wave * 4 + kt) * 32 + lane) * 16];

#pragma unroll
    for (int nt = 0; nt < NT2; ++nt) {
        v8f acc = {};
#pragma unroll
        for (int kt = 0; kt < 4; ++kt) {
            Frag B;
            B.v = *(const v16bf*)&sW2[((kt * NT2 + nt) * 32 + lane) * 16];
            acc = __builtin_amdgcn_wmma_f32_16x16x32_bf16(
                false, A2[kt].v, false, B.v, (short)0, acc, false, false);
        }
        int col  = nt * 16 + cLo;
        float bb = sB2[col];
#pragma unroll
        for (int r = 0; r < 8; ++r) {
            int M   = mbase + r;
            float z = acc[r] + bb;
            if constexpr (!LAST) {
                int row = m0 + M;
                if (row < NN) out[(size_t)row * CH + col] = fmaxf(z, 0.f);
            } else {
                if (col < DOUT) sLogit[(wave * 16 + M) * 48 + col] = z;
            }
        }
    }

    if constexpr (LAST) {
        __syncthreads();
        if (tid < 128) {
            int row = blockIdx.x * 128 + tid;
            if (row < NN) {
                const float* l = &sLogit[tid * 48];
                float mx = -3.4e38f;
                for (int c = 0; c < DOUT; ++c) mx = fmaxf(mx, l[c]);
                float s = 0.f;
                for (int c = 0; c < DOUT; ++c) s += __expf(l[c] - mx);
                float ls = __logf(s) + mx;
                float* o = out + (size_t)row * DOUT;
                for (int c = 0; c < DOUT; ++c) o[c] = l[c] - ls;
            }
        }
    }
}

// ---------------------------------------------------------------- host launcher
extern "C" void kernel_launch(void* const* d_in, const int* in_sizes, int n_in,
                              void* d_out, int out_size, void* d_ws, size_t ws_size,
                              hipStream_t stream) {
    // Identify inputs by flat size; remaining leaves are the 24 params, in order.
    const float* x  = nullptr;
    const int*   ei = nullptr;
    int pidx[24];
    int np = 0;
    for (int i = 0; i < n_in; ++i) {
        if (in_sizes[i] == NN * CH && !x)       x  = (const float*)d_in[i];
        else if (in_sizes[i] == 2 * NE && !ei)  ei = (const int*)d_in[i];
        else if (np < 24)                       pidx[np++] = i;
    }

    const float *W1[3], *B1[3], *GA[3], *BE[3], *MN[3], *VR[3], *W2[3], *B2[3];
    for (int l = 0; l < 3; ++l) {
        int b = 8 * l;
        const float* p[8];
        for (int k = 0; k < 8; ++k) p[k] = (const float*)d_in[pidx[b + k]];
        if (in_sizes[pidx[b]] == 128 * 128) {
            // insertion order: lin1.W, lin1.b, gamma, beta, mean, var, lin2.W, lin2.b
            W1[l]=p[0]; B1[l]=p[1]; GA[l]=p[2]; BE[l]=p[3]; MN[l]=p[4]; VR[l]=p[5]; W2[l]=p[6]; B2[l]=p[7];
        } else {
            // key-sorted order: beta, gamma, mean, var, lin1.W, lin1.b, lin2.W, lin2.b
            BE[l]=p[0]; GA[l]=p[1]; MN[l]=p[2]; VR[l]=p[3]; W1[l]=p[4]; B1[l]=p[5]; W2[l]=p[6]; B2[l]=p[7];
        }
    }

    float* bufA = (float*)d_ws;                       // [NN][128] h_in (x + agg)
    float* bufB = bufA + (size_t)NN * CH;             // [NN][128] layer output

    const int copyBlocks = (NN * CH / 4 + 255) / 256;
    const int scatBlocks = (int)(((long long)NE * 32 + 255) / 256);
    const int mlpBlocks  = (NN + 127) / 128;

    const float* cur = x;
    for (int l = 0; l < 3; ++l) {
        gin_copy4<<<copyBlocks, 256, 0, stream>>>((const float4*)cur, (float4*)bufA, NN * CH / 4);
        gin_scatter<<<scatBlocks, 256, 0, stream>>>(cur, ei, bufA);
        if (l < 2) {
            gin_mlp<128, false><<<mlpBlocks, 256, 0, stream>>>(
                bufA, W1[l], B1[l], GA[l], BE[l], MN[l], VR[l], W2[l], B2[l], bufB);
            cur = bufB;
        } else {
            gin_mlp<47, true><<<mlpBlocks, 256, 0, stream>>>(
                bufA, W1[l], B1[l], GA[l], BE[l], MN[l], VR[l], W2[l], B2[l], (float*)d_out);
        }
    }
}
